// SpatialAttention_9663676416297
// MI455X (gfx1250) — matrix operand order repaired and checked by reading, not compiled, not run on hardware
//
#include <hip/hip_runtime.h>
#include <hip/hip_bf16.h>
#include <stdint.h>

#define NUM_HEADS 6
#define NTOK 256
#define QK_STRIDE 72      // bf16 elems per Q/K row: 64 + 8 pad (keeps 16B align, spreads banks)
#define VT_STRIDE 264     // bf16 elems per Vt/P row: 256 + 8 pad

typedef __bf16 v16bf __attribute__((ext_vector_type(16)));
typedef float  v8f   __attribute__((ext_vector_type(8)));
typedef float  v4f   __attribute__((ext_vector_type(4)));   // native vector for nontemporal builtins

static __device__ __forceinline__ uint32_t pk_bf16(float lo, float hi) {
    __bf16 a = (__bf16)lo, b = (__bf16)hi;
    uint16_t ua = __builtin_bit_cast(uint16_t, a);
    uint16_t ub = __builtin_bit_cast(uint16_t, b);
    return (uint32_t)ua | ((uint32_t)ub << 16);
}

// A-matrix fragment (16x32 bf16): lane r=L&15 holds K in {k0+hi*8 .. +8} and {+16..}
static __device__ __forceinline__ v16bf load_frag_a(const __bf16* base, int row0, int k0,
                                                    int stride, int lane) {
    int r  = row0 + (lane & 15);
    int kk = k0 + ((lane >> 4) << 3);
    const __bf16* p = base + r * stride + kk;
    union { uint4 u[2]; v16bf v; } t;
    t.u[0] = *(const uint4*)(p);
    t.u[1] = *(const uint4*)(p + 16);
    return t.v;
}

// B-matrix fragment (32x16 bf16): lane c=L&15 holds K in {k0+hi*8 .. +8} and {+16..}
static __device__ __forceinline__ v16bf load_frag_b(const __bf16* base, int col0, int k0,
                                                    int stride, int lane) {
    int c  = col0 + (lane & 15);
    int kk = k0 + ((lane >> 4) << 3);
    const __bf16* p = base + c * stride + kk;
    union { uint4 u[2]; v16bf v; } t;
    t.u[0] = *(const uint4*)(p);
    t.u[1] = *(const uint4*)(p + 16);
    return t.v;
}

// ---------------- DynamicPosBias MLP: 961 rows -> [961,6] table ----------------
static __device__ void ln_relu(float* x, int n, const float* g, const float* b) {
    float m = 0.f;
    for (int j = 0; j < n; ++j) m += x[j];
    m /= (float)n;
    float v = 0.f;
    for (int j = 0; j < n; ++j) { float d = x[j] - m; v += d * d; }
    v /= (float)n;
    float inv = rsqrtf(v + 1e-3f);
    for (int j = 0; j < n; ++j) {
        float t = (x[j] - m) * inv * g[j] + b[j];
        x[j] = t > 0.f ? t : 0.f;
    }
}

__global__ void pos_mlp_kernel(const float* __restrict__ w0, const float* __restrict__ b0,
                               const float* __restrict__ g1, const float* __restrict__ bb1,
                               const float* __restrict__ d1w, const float* __restrict__ d1b,
                               const float* __restrict__ g2, const float* __restrict__ bb2,
                               const float* __restrict__ d2w, const float* __restrict__ d2b,
                               const float* __restrict__ g3, const float* __restrict__ bb3,
                               const float* __restrict__ d3w, const float* __restrict__ d3b,
                               float* __restrict__ posTab) {
    int i = blockIdx.x * blockDim.x + threadIdx.x;
    if (i >= 961) return;
    float dh = (float)(i / 31 - 15);
    float dw = (float)(i % 31 - 15);
    float x[48];
    for (int j = 0; j < 48; ++j) x[j] = dh * w0[j] + dw * w0[48 + j] + b0[j];
    ln_relu(x, 48, g1, bb1);
    float y[24];
    for (int j = 0; j < 24; ++j) {
        float s = d1b[j];
        for (int k = 0; k < 48; ++k) s += x[k] * d1w[k * 24 + j];
        y[j] = s;
    }
    ln_relu(y, 24, g2, bb2);
    float z[24];
    for (int j = 0; j < 24; ++j) {
        float s = d2b[j];
        for (int k = 0; k < 24; ++k) s += y[k] * d2w[k * 24 + j];
        z[j] = s;
    }
    ln_relu(z, 24, g3, bb3);
    for (int h = 0; h < NUM_HEADS; ++h) {
        float s = d3b[h];
        for (int k = 0; k < 24; ++k) s += z[k] * d3w[k * NUM_HEADS + h];
        posTab[i * NUM_HEADS + h] = s;
    }
}

// ---------------- Windowed attention: 1 block per (window, head) ----------------
__global__ __launch_bounds__(256, 1) void attn_kernel(const float* __restrict__ qkv,
                                                      const float* __restrict__ posTab,
                                                      float* __restrict__ out) {
    __shared__ __align__(16) __bf16 sQ[NTOK * QK_STRIDE];
    __shared__ __align__(16) __bf16 sK[NTOK * QK_STRIDE];
    __shared__ __align__(16) __bf16 sVt[64 * VT_STRIDE];
    __shared__ __align__(16) __bf16 sP[8 * 16 * VT_STRIDE];
    __shared__ float sPos[961];

    const int tid  = threadIdx.x;
    const int lane = tid & 31;
    const int wid  = tid >> 5;
    const int head = blockIdx.x % NUM_HEADS;
    const int win  = blockIdx.x / NUM_HEADS;
    const int b  = win >> 4;
    const int wh = (win >> 2) & 3;
    const int ww = win & 3;

    const long planeStride = (long)16 * 4096 * 384;
    const long base = (long)b * 4096 * 384 + head * 64;
    const float* Qg = qkv + base;
    const float* Kg = qkv + planeStride + base;
    const float* Vg = qkv + 2 * planeStride + base;

    // ---- stage Q (scaled), K row-major, V transposed -> LDS bf16 ----
    // Streaming data (read exactly once across the whole grid): non-temporal loads.
    const int rl = tid >> 4;   // row-in-chunk 0..15
    const int c4 = tid & 15;   // float4 column 0..15
    for (int it = 0; it < 16; ++it) {
        int q = it * 16 + rl;
        int L = ((wh << 4) + (q >> 4)) * 64 + (ww << 4) + (q & 15);
        long off = (long)L * 384 + c4 * 4;
        v4f qv = __builtin_nontemporal_load((const v4f*)(Qg + off));
        v4f kv = __builtin_nontemporal_load((const v4f*)(Kg + off));
        v4f vv = __builtin_nontemporal_load((const v4f*)(Vg + off));
        const float scale = 0.125f;  // 64^-0.5
        uint2 qp, kp;
        qp.x = pk_bf16(qv[0] * scale, qv[1] * scale);
        qp.y = pk_bf16(qv[2] * scale, qv[3] * scale);
        kp.x = pk_bf16(kv[0], kv[1]);
        kp.y = pk_bf16(kv[2], kv[3]);
        *(uint2*)&sQ[q * QK_STRIDE + c4 * 4] = qp;
        *(uint2*)&sK[q * QK_STRIDE + c4 * 4] = kp;
        sVt[(c4 * 4 + 0) * VT_STRIDE + q] = (__bf16)vv[0];
        sVt[(c4 * 4 + 1) * VT_STRIDE + q] = (__bf16)vv[1];
        sVt[(c4 * 4 + 2) * VT_STRIDE + q] = (__bf16)vv[2];
        sVt[(c4 * 4 + 3) * VT_STRIDE + q] = (__bf16)vv[3];
    }
    for (int i = tid; i < 961; i += 256) sPos[i] = posTab[i * NUM_HEADS + head];
    __syncthreads();

    const int ln16 = lane & 15;
    const int hi8  = (lane >> 4) << 3;
    __bf16* pw = sP + wid * 16 * VT_STRIDE;
    const v8f vzero = {0.f, 0.f, 0.f, 0.f, 0.f, 0.f, 0.f, 0.f};

    for (int itq = 0; itq < 2; ++itq) {
        const int qt = itq * 8 + wid;  // query tile 0..15
        v16bf qf0 = load_frag_a(sQ, qt * 16, 0, QK_STRIDE, lane);
        v16bf qf1 = load_frag_a(sQ, qt * 16, 32, QK_STRIDE, lane);

        // S = Q * K^T  (16 tiles of 16x16, fp32 accum)
        v8f S[16];
#pragma unroll
        for (int kt = 0; kt < 16; ++kt) {
            v16bf kf0 = load_frag_b(sK, kt * 16, 0, QK_STRIDE, lane);
            v16bf kf1 = load_frag_b(sK, kt * 16, 32, QK_STRIDE, lane);
            v8f c = vzero;
            c = __builtin_amdgcn_wmma_f32_16x16x32_bf16(false, qf0, false, kf0, (short)0, c, false, false);
            c = __builtin_amdgcn_wmma_f32_16x16x32_bf16(false, qf1, false, kf1, (short)0, c, false, false);
            S[kt] = c;
        }

        // + relative position bias (analytic index into 961-entry table)
#pragma unroll
        for (int kt = 0; kt < 16; ++kt) {
            int n  = kt * 16 + ln16;
            int kh = n >> 4, kw = n & 15;
#pragma unroll
            for (int j = 0; j < 8; ++j) {
                int m = qt * 16 + j + hi8;
                int idx = ((m >> 4) - kh + 15) * 31 + ((m & 15) - kw + 15);
                S[kt][j] += sPos[idx];
            }
        }

        // softmax over each row (row lives in one 16-lane half per VGPR slot)
#pragma unroll
        for (int j = 0; j < 8; ++j) {
            float mx = -3.4e38f;
#pragma unroll
            for (int kt = 0; kt < 16; ++kt) mx = fmaxf(mx, S[kt][j]);
            for (int d = 1; d < 16; d <<= 1) mx = fmaxf(mx, __shfl_xor(mx, d, 32));
            float sum = 0.f;
#pragma unroll
            for (int kt = 0; kt < 16; ++kt) {
                float e = __expf(S[kt][j] - mx);
                S[kt][j] = e;
                sum += e;
            }
            for (int d = 1; d < 16; d <<= 1) sum += __shfl_xor(sum, d, 32);
            float inv = 1.0f / sum;
#pragma unroll
            for (int kt = 0; kt < 16; ++kt) S[kt][j] *= inv;
        }

        // P -> LDS (C-layout -> A-layout via shared scratch), bf16
#pragma unroll
        for (int kt = 0; kt < 16; ++kt)
#pragma unroll
            for (int j = 0; j < 8; ++j)
                pw[(j + hi8) * VT_STRIDE + kt * 16 + ln16] = (__bf16)S[kt][j];
        __syncthreads();

        // O = P * V  (4 output tiles of 16x16, accumulate over 8 k-chunks)
        v8f O[4] = {vzero, vzero, vzero, vzero};
#pragma unroll
        for (int kb = 0; kb < 8; ++kb) {
            v16bf pf = load_frag_a(pw, 0, kb * 32, VT_STRIDE, lane);
#pragma unroll
            for (int n = 0; n < 4; ++n) {
                v16bf vf = load_frag_b(sVt, n * 16, kb * 32, VT_STRIDE, lane);
                O[n] = __builtin_amdgcn_wmma_f32_16x16x32_bf16(false, pf, false, vf, (short)0, O[n], false, false);
            }
        }

        // store: out[b, h, w, head*64 + d]  (write-once stream: non-temporal)
#pragma unroll
        for (int n = 0; n < 4; ++n) {
            int d = n * 16 + ln16;
#pragma unroll
            for (int j = 0; j < 8; ++j) {
                int q = qt * 16 + j + hi8;
                int h = (wh << 4) + (q >> 4);
                int w = (ww << 4) + (q & 15);
                long o = (((long)b * 64 + h) * 64 + w) * 384 + head * 64 + d;
                __builtin_nontemporal_store(O[n][j], out + o);
            }
        }
        __syncthreads();  // protect P scratch before next iteration rewrites it
    }
}

extern "C" void kernel_launch(void* const* d_in, const int* in_sizes, int n_in,
                              void* d_out, int out_size, void* d_ws, size_t ws_size,
                              hipStream_t stream) {
    const float* qkv = (const float*)d_in[0];
    const float* pos_proj_w = (const float*)d_in[1];
    const float* pos_proj_b = (const float*)d_in[2];
    const float* ln1_g = (const float*)d_in[3];
    const float* ln1_b = (const float*)d_in[4];
    const float* d1_w = (const float*)d_in[5];
    const float* d1_b = (const float*)d_in[6];
    const float* ln2_g = (const float*)d_in[7];
    const float* ln2_b = (const float*)d_in[8];
    const float* d2_w = (const float*)d_in[9];
    const float* d2_b = (const float*)d_in[10];
    const float* ln3_g = (const float*)d_in[11];
    const float* ln3_b = (const float*)d_in[12];
    const float* d3_w = (const float*)d_in[13];
    const float* d3_b = (const float*)d_in[14];
    float* out = (float*)d_out;
    float* posTab = (float*)d_ws;  // 961*6 floats

    pos_mlp_kernel<<<4, 256, 0, stream>>>(pos_proj_w, pos_proj_b, ln1_g, ln1_b,
                                          d1_w, d1_b, ln2_g, ln2_b, d2_w, d2_b,
                                          ln3_g, ln3_b, d3_w, d3_b, posTab);
    attn_kernel<<<256 * NUM_HEADS, 256, 0, stream>>>(qkv, posTab, out);
}
